// CausalSelfAttention_39273180955237
// MI455X (gfx1250) — matrix operand order repaired and checked by reading, not compiled, not run on hardware
//
#include <hip/hip_runtime.h>
#include <hip/hip_bf16.h>
#include <stdint.h>

#define B_ 2
#define T_ 2048
#define C_ 1024
#define H_ 16
#define D_ 64
#define XSTR 40   // LDS row stride in bf16 elems: 80B -> 16B-aligned rows, conflict-free
#define NKSTEP (C_ / 32)

typedef __bf16 bf16_t;
typedef bf16_t v16bf __attribute__((ext_vector_type(16)));
typedef bf16_t v8bf  __attribute__((ext_vector_type(8)));
typedef bf16_t v4bf  __attribute__((ext_vector_type(4)));
typedef float  v8f   __attribute__((ext_vector_type(8)));

static __device__ inline v8f vzero8() { v8f z = {0,0,0,0,0,0,0,0}; return z; }

static __device__ inline v16bf cat16(v8bf lo, v8bf hi) {
  return __builtin_shufflevector(lo, hi, 0,1,2,3,4,5,6,7,8,9,10,11,12,13,14,15);
}

// A-fragment (16x32 bf16, MxK). Lane: m = lane&15, half = lane>>4.
// elems 0..7 = K[half*8 .. +7], elems 8..15 = K[16+half*8 .. +7]
static __device__ inline v16bf load_afrag(const bf16_t* rowp, int half) {
  v8bf lo = *(const v8bf*)(rowp + half * 8);
  v8bf hi = *(const v8bf*)(rowp + 16 + half * 8);
  return cat16(lo, hi);
}

// B-fragment (32x16 bf16, KxN). Lane: n = lane&15, half = lane>>4.
// elems 0..7 = K[half*8 .. +7], elems 8..15 = K[16+half*8 .. +7] for column n.
static __device__ inline v16bf load_bfrag(const bf16_t* rowp, int half) {
  const bf16_t* p = rowp + half * 8;
  return cat16(*(const v8bf*)p, *(const v8bf*)(p + 16));
}

static __device__ inline v8f wmma_bf16(v16bf a, v16bf b, v8f c) {
  return __builtin_amdgcn_wmma_f32_16x16x32_bf16(false, a, false, b, (short)0, c,
                                                 false, false);
}

// ---------------------------------------------------------------------------
// Kernel 0: fp32 -> bf16 pre-conversion (x, Wqkv, Wproj).
// ---------------------------------------------------------------------------
__global__ __launch_bounds__(256) void cvt_kernel(const float* __restrict__ src,
                                                  bf16_t* __restrict__ dst, int n4) {
  const int i = blockIdx.x * 256 + threadIdx.x;
  if (i < n4) {
    const float4 f = ((const float4*)src)[i];
    v4bf v; v[0] = (bf16_t)f.x; v[1] = (bf16_t)f.y; v[2] = (bf16_t)f.z; v[3] = (bf16_t)f.w;
    ((v4bf*)dst)[i] = v;
  }
}

// ---------------------------------------------------------------------------
// WMMA GEMM mainloop on bf16: 128x128 block tile = X[128,1024] @ W[128,1024]^T.
// 256 threads = 8 waves; wave (rw = (wv&3)*32 rows, cw = (wv>>2)*64 cols) does
// 2 A-frags x 4 B-frags = 8 WMMAs per K-step. Register + LDS double buffering
// with an XOR-toggled buffer offset and unroll(1) so the allocator keeps the
// 8 accumulator tuples pinned (no v_mov shuffling). One barrier per K-step;
// global loads of kt+1 overlap the WMMAs of kt.
// ---------------------------------------------------------------------------
static __device__ inline void gemm_mainloop(const bf16_t* __restrict__ X,
                                            const bf16_t* __restrict__ W,
                                            int m0, int n0,
                                            bf16_t* Xs, bf16_t* Ws,  // [2][128*XSTR]
                                            v8f acc[2][4]) {
  const int tid  = threadIdx.x;
  const int wv   = tid >> 5;
  const int lane = tid & 31;
  const int col  = lane & 15;
  const int half = lane >> 4;
  const int rw   = (wv & 3) * 32;
  const int cw   = (wv >> 2) * 64;
#pragma unroll
  for (int i = 0; i < 2; ++i)
#pragma unroll
    for (int j = 0; j < 4; ++j) acc[i][j] = vzero8();

  // Staging: 16B chunks; X tile 128x32 and W tile 128x32 -> 2 chunks each per thread.
  const int srow = tid >> 2;           // 0..63
  const int skc  = (tid & 3) * 8;
  const bf16_t* Xg0 = X + (size_t)(m0 + srow) * C_ + skc;
  const bf16_t* Xg1 = X + (size_t)(m0 + 64 + srow) * C_ + skc;
  const bf16_t* Wg0 = W + (size_t)(n0 + srow) * C_ + skc;
  const bf16_t* Wg1 = W + (size_t)(n0 + 64 + srow) * C_ + skc;
  bf16_t* xs0 = Xs + srow * XSTR + skc;
  bf16_t* xs1 = Xs + (64 + srow) * XSTR + skc;
  bf16_t* ws0 = Ws + srow * XSTR + skc;
  bf16_t* ws1 = Ws + (64 + srow) * XSTR + skc;

  // Preload K-step 0 into buffer 0.
  v8bf xa = *(const v8bf*)Xg0;
  v8bf xb = *(const v8bf*)Xg1;
  v8bf wa = *(const v8bf*)Wg0;
  v8bf wb = *(const v8bf*)Wg1;
  *(v8bf*)xs0 = xa; *(v8bf*)xs1 = xb;
  *(v8bf*)ws0 = wa; *(v8bf*)ws1 = wb;
  __syncthreads();

  const unsigned BUF = 128 * XSTR;     // elements per buffer
  unsigned co = 0;                     // compute-buffer offset (toggles)
#pragma unroll 1
  for (int kt = 0; kt < NKSTEP; ++kt) {
    if (kt + 1 < NKSTEP) {             // (a) prefetch kt+1 into registers
      xa = *(const v8bf*)(Xg0 + (kt + 1) * 32);
      xb = *(const v8bf*)(Xg1 + (kt + 1) * 32);
      wa = *(const v8bf*)(Wg0 + (kt + 1) * 32);
      wb = *(const v8bf*)(Wg1 + (kt + 1) * 32);
    }
    // (b) compute on buffer co: 2 A-frags, 4 B-frags, 8 WMMAs
    const bf16_t* xsb = Xs + co;
    const bf16_t* wsb = Ws + co;
    v16bf a0 = load_afrag(xsb + (rw + col) * XSTR, half);
    v16bf a1 = load_afrag(xsb + (rw + 16 + col) * XSTR, half);
    v16bf b0 = load_bfrag(wsb + (cw + col) * XSTR, half);
    v16bf b1 = load_bfrag(wsb + (cw + 16 + col) * XSTR, half);
    v16bf b2 = load_bfrag(wsb + (cw + 32 + col) * XSTR, half);
    v16bf b3 = load_bfrag(wsb + (cw + 48 + col) * XSTR, half);
    acc[0][0] = wmma_bf16(a0, b0, acc[0][0]);
    acc[0][1] = wmma_bf16(a0, b1, acc[0][1]);
    acc[0][2] = wmma_bf16(a0, b2, acc[0][2]);
    acc[0][3] = wmma_bf16(a0, b3, acc[0][3]);
    acc[1][0] = wmma_bf16(a1, b0, acc[1][0]);
    acc[1][1] = wmma_bf16(a1, b1, acc[1][1]);
    acc[1][2] = wmma_bf16(a1, b2, acc[1][2]);
    acc[1][3] = wmma_bf16(a1, b3, acc[1][3]);
    // (c) store prefetched regs into the other buffer (its last readers were
    //     fenced by the previous iteration's barrier), then a single barrier.
    if (kt + 1 < NKSTEP) {
      const unsigned no = co ^ BUF;
      *(v8bf*)(xs0 + no) = xa;
      *(v8bf*)(xs1 + no) = xb;
      *(v8bf*)(ws0 + no) = wa;
      *(v8bf*)(ws1 + no) = wb;
      __syncthreads();
    }
    co ^= BUF;
  }
}

// ---------------------------------------------------------------------------
// Kernel 1: QKV projection. Grid (24, 32). Scatters Q (pre-scaled), K as
// (B,H,T,D) bf16 and V transposed as (B,H,D,T) bf16.
// ---------------------------------------------------------------------------
__global__ __launch_bounds__(256) void qkv_kernel(const bf16_t* __restrict__ xb,
                                                  const bf16_t* __restrict__ Wqkv,
                                                  const float* __restrict__ bqkv,
                                                  bf16_t* __restrict__ qws,
                                                  bf16_t* __restrict__ kws,
                                                  bf16_t* __restrict__ vtws) {
  __shared__ alignas(16) bf16_t Xs[2 * 128 * XSTR];
  __shared__ alignas(16) bf16_t Ws[2 * 128 * XSTR];
  const int n0 = blockIdx.x * 128, m0 = blockIdx.y * 128;
  v8f acc[2][4];
  gemm_mainloop(xb, Wqkv, m0, n0, Xs, Ws, acc);

  const int tid = threadIdx.x, wv = tid >> 5, lane = tid & 31;
  const int col = lane & 15, half = lane >> 4;
  const int rw = (wv & 3) * 32, cw = (wv >> 2) * 64;
  const int nb = n0 + cw;               // 64-col group = exactly one head
  const int which = nb >> 10;           // 0=Q, 1=K, 2=V
  const int h = (nb & (C_ - 1)) >> 6;
#pragma unroll
  for (int j = 0; j < 4; ++j) {
    const int d = j * 16 + col;
    const float bias = bqkv[nb + d];
#pragma unroll
    for (int i = 0; i < 2; ++i) {
#pragma unroll
      for (int r = 0; r < 8; ++r) {
        const int m = m0 + rw + i * 16 + r + 8 * half;
        const int bb = m >> 11;
        const int t = m & (T_ - 1);
        const float val = acc[i][j][r] + bias;
        if (which == 0)
          qws[((size_t)(bb * H_ + h) * T_ + t) * D_ + d] = (bf16_t)(val * 0.125f);
        else if (which == 1)
          kws[((size_t)(bb * H_ + h) * T_ + t) * D_ + d] = (bf16_t)val;
        else
          vtws[((size_t)(bb * H_ + h) * D_ + d) * T_ + t] = (bf16_t)val;
      }
    }
  }
}

// ---------------------------------------------------------------------------
// Kernel 2: flash attention, one wave per 32-row Q tile. Grid (T/32, B*H).
// Each K/V fragment feeds 2 WMMAs (row groups i=0,1). Writes O as bf16 in
// (B*T, C) layout for the out-projection GEMM.
// ---------------------------------------------------------------------------
__global__ __launch_bounds__(32) void attn_kernel(const bf16_t* __restrict__ qws,
                                                  const bf16_t* __restrict__ kws,
                                                  const bf16_t* __restrict__ vtws,
                                                  const unsigned char* __restrict__ pad,
                                                  bf16_t* __restrict__ obf) {
  __shared__ alignas(16) bf16_t Pls[32 * XSTR];
  const int lane = threadIdx.x, col = lane & 15, half = lane >> 4;
  const int q0 = blockIdx.x * 32;
  const int bh = blockIdx.y;
  const int b = bh >> 4, h = bh & (H_ - 1);
  const bf16_t* Qb = qws + ((size_t)bh * T_ + q0) * D_;
  const bf16_t* Kb = kws + (size_t)bh * T_ * D_;
  const bf16_t* Vt = vtws + (size_t)bh * D_ * T_;
  const unsigned char* pm = pad + (size_t)b * T_;

  v16bf qf[2][2];                       // [row group][d half]
#pragma unroll
  for (int i = 0; i < 2; ++i)
#pragma unroll
    for (int s = 0; s < 2; ++s)
      qf[i][s] = load_afrag(Qb + (size_t)(i * 16 + col) * D_ + s * 32, half);

  float mrun[2][8], lrun[2][8], corr[2][8];
  v8f oacc[2][4];
#pragma unroll
  for (int i = 0; i < 2; ++i) {
#pragma unroll
    for (int r = 0; r < 8; ++r) { mrun[i][r] = -3.0e38f; lrun[i][r] = 0.0f; }
#pragma unroll
    for (int j = 0; j < 4; ++j) oacc[i][j] = vzero8();
  }

  const int kend = q0 + 32;
#pragma unroll 1
  for (int k0 = 0; k0 < kend; k0 += 32) {
    // Scores: S[32q x 32k]. 4 K fragments feed 8 WMMAs.
    v8f sc[2][2];
#pragma unroll
    for (int i = 0; i < 2; ++i) { sc[i][0] = vzero8(); sc[i][1] = vzero8(); }
    {
      v16bf b00 = load_bfrag(Kb + (size_t)(k0 + col) * D_, half);        // t=0, d 0..31
      v16bf b01 = load_bfrag(Kb + (size_t)(k0 + col) * D_ + 32, half);   // t=0, d 32..63
      v16bf b10 = load_bfrag(Kb + (size_t)(k0 + 16 + col) * D_, half);   // t=1
      v16bf b11 = load_bfrag(Kb + (size_t)(k0 + 16 + col) * D_ + 32, half);
#pragma unroll
      for (int i = 0; i < 2; ++i) {
        sc[i][0] = wmma_bf16(qf[i][0], b00, sc[i][0]);
        sc[i][0] = wmma_bf16(qf[i][1], b01, sc[i][0]);
        sc[i][1] = wmma_bf16(qf[i][0], b10, sc[i][1]);
        sc[i][1] = wmma_bf16(qf[i][1], b11, sc[i][1]);
      }
    }
    const int key0 = k0 + col, key1 = k0 + 16 + col;
    const bool ok0 = pm[key0] != 0, ok1 = pm[key1] != 0;

    // prior-iteration P-fragment reads must land before we overwrite Pls
    asm volatile("s_wait_dscnt 0x0" ::: "memory");

#pragma unroll
    for (int i = 0; i < 2; ++i) {
#pragma unroll
      for (int r = 0; r < 8; ++r) {
        const int rowq = q0 + i * 16 + r + 8 * half;
        const bool v0 = ok0 && (key0 <= rowq);
        const bool v1 = ok1 && (key1 <= rowq);
        float e0 = v0 ? sc[i][0][r] : -3.0e38f;
        float e1 = v1 ? sc[i][1][r] : -3.0e38f;
        float mx = fmaxf(e0, e1);
#pragma unroll
        for (int off = 1; off < 16; off <<= 1) mx = fmaxf(mx, __shfl_xor(mx, off, 32));
        const float mn = fmaxf(mrun[i][r], mx);
        const float c = __expf(mrun[i][r] - mn);
        const float p0 = v0 ? __expf(e0 - mn) : 0.0f;
        const float p1 = v1 ? __expf(e1 - mn) : 0.0f;
        float rs = p0 + p1;
#pragma unroll
        for (int off = 1; off < 16; off <<= 1) rs += __shfl_xor(rs, off, 32);
        lrun[i][r] = lrun[i][r] * c + rs;
        mrun[i][r] = mn;
        corr[i][r] = c;
        Pls[(i * 16 + r + 8 * half) * XSTR + col] = (bf16_t)p0;
        Pls[(i * 16 + r + 8 * half) * XSTR + 16 + col] = (bf16_t)p1;
      }
#pragma unroll
      for (int j = 0; j < 4; ++j) {
#pragma unroll
        for (int r = 0; r < 8; ++r) oacc[i][j][r] *= corr[i][r];
      }
    }
    // make the cross-lane P transpose in LDS visible before fragment reads
    asm volatile("s_wait_dscnt 0x0" ::: "memory");

    v16bf pa0 = load_afrag(Pls + (col) * XSTR, half);        // rows 0..15
    v16bf pa1 = load_afrag(Pls + (16 + col) * XSTR, half);   // rows 16..31
    v16bf vf0 = load_bfrag(Vt + (size_t)(0 * 16 + col) * T_ + k0, half);
    v16bf vf1 = load_bfrag(Vt + (size_t)(1 * 16 + col) * T_ + k0, half);
    v16bf vf2 = load_bfrag(Vt + (size_t)(2 * 16 + col) * T_ + k0, half);
    v16bf vf3 = load_bfrag(Vt + (size_t)(3 * 16 + col) * T_ + k0, half);
    oacc[0][0] = wmma_bf16(pa0, vf0, oacc[0][0]);
    oacc[0][1] = wmma_bf16(pa0, vf1, oacc[0][1]);
    oacc[0][2] = wmma_bf16(pa0, vf2, oacc[0][2]);
    oacc[0][3] = wmma_bf16(pa0, vf3, oacc[0][3]);
    oacc[1][0] = wmma_bf16(pa1, vf0, oacc[1][0]);
    oacc[1][1] = wmma_bf16(pa1, vf1, oacc[1][1]);
    oacc[1][2] = wmma_bf16(pa1, vf2, oacc[1][2]);
    oacc[1][3] = wmma_bf16(pa1, vf3, oacc[1][3]);
  }

#pragma unroll
  for (int i = 0; i < 2; ++i) {
#pragma unroll
    for (int r = 0; r < 8; ++r) lrun[i][r] = 1.0f / lrun[i][r];
#pragma unroll
    for (int j = 0; j < 4; ++j) {
#pragma unroll
      for (int r = 0; r < 8; ++r) {
        const size_t m = (size_t)b * T_ + q0 + i * 16 + r + 8 * half;
        obf[m * C_ + h * D_ + j * 16 + col] = (bf16_t)(oacc[i][j][r] * lrun[i][r]);
      }
    }
  }
}

// ---------------------------------------------------------------------------
// Kernel 3: output projection from bf16 O. Grid (8, 32). fp32 result.
// ---------------------------------------------------------------------------
__global__ __launch_bounds__(256) void proj_kernel(const bf16_t* __restrict__ ob,
                                                   const bf16_t* __restrict__ Wp,
                                                   const float* __restrict__ bp,
                                                   float* __restrict__ out) {
  __shared__ alignas(16) bf16_t Xs[2 * 128 * XSTR];
  __shared__ alignas(16) bf16_t Ws[2 * 128 * XSTR];
  const int n0 = blockIdx.x * 128, m0 = blockIdx.y * 128;
  v8f acc[2][4];
  gemm_mainloop(ob, Wp, m0, n0, Xs, Ws, acc);

  const int tid = threadIdx.x, wv = tid >> 5, lane = tid & 31;
  const int col = lane & 15, half = lane >> 4;
  const int rw = (wv & 3) * 32, cw = (wv >> 2) * 64;
  const int nb = n0 + cw;
#pragma unroll
  for (int j = 0; j < 4; ++j) {
    const float bias = bp[nb + j * 16 + col];
#pragma unroll
    for (int i = 0; i < 2; ++i) {
#pragma unroll
      for (int r = 0; r < 8; ++r) {
        const size_t m = (size_t)m0 + rw + i * 16 + r + 8 * half;
        out[m * C_ + nb + j * 16 + col] = acc[i][j][r] + bias;
      }
    }
  }
}

// ---------------------------------------------------------------------------
extern "C" void kernel_launch(void* const* d_in, const int* in_sizes, int n_in,
                              void* d_out, int out_size, void* d_ws, size_t ws_size,
                              hipStream_t stream) {
  (void)in_sizes; (void)n_in; (void)out_size; (void)ws_size;
  const float* x           = (const float*)d_in[0];
  const unsigned char* pad = (const unsigned char*)d_in[1];  // bool mask (B,T)
  const float* Wqkv        = (const float*)d_in[2];
  const float* bqkv        = (const float*)d_in[3];
  const float* Wproj       = (const float*)d_in[4];
  const float* bproj       = (const float*)d_in[5];
  float* out = (float*)d_out;

  char* ws = (char*)d_ws;
  const size_t nBTC = (size_t)B_ * T_ * C_;        // 4M elements
  bf16_t* qws  = (bf16_t*)(ws);                    //  8 MB
  bf16_t* kws  = (bf16_t*)(ws + nBTC * 2);         //  8 MB
  bf16_t* vtws = (bf16_t*)(ws + nBTC * 4);         //  8 MB (transposed V)
  bf16_t* obf  = (bf16_t*)(ws + nBTC * 6);         //  8 MB (attention output, bf16)
  bf16_t* xbf  = (bf16_t*)(ws + nBTC * 8);         //  8 MB
  bf16_t* wqb  = (bf16_t*)(ws + nBTC * 10);        //  6 MB
  bf16_t* wpb  = (bf16_t*)(ws + nBTC * 10 + (size_t)3 * C_ * C_ * 2);  // 2 MB

  const int nx4 = (int)(nBTC / 4);
  const int nq4 = 3 * C_ * C_ / 4;
  const int np4 = C_ * C_ / 4;
  cvt_kernel<<<dim3((nx4 + 255) / 256), dim3(256), 0, stream>>>(x, xbf, nx4);
  cvt_kernel<<<dim3((nq4 + 255) / 256), dim3(256), 0, stream>>>(Wqkv, wqb, nq4);
  cvt_kernel<<<dim3((np4 + 255) / 256), dim3(256), 0, stream>>>(Wproj, wpb, np4);

  qkv_kernel<<<dim3(24, 32, 1), dim3(256, 1, 1), 0, stream>>>(xbf, wqb, bqkv,
                                                              qws, kws, vtws);
  attn_kernel<<<dim3(T_ / 32, B_ * H_, 1), dim3(32, 1, 1), 0, stream>>>(
      qws, kws, vtws, pad, obf);
  proj_kernel<<<dim3(8, 32, 1), dim3(256, 1, 1), 0, stream>>>(obf, wpb, bproj, out);
}
